// LN_MHA_60739427500338
// MI455X (gfx1250) — matrix operand order repaired and checked by reading, not compiled, not run on hardware
//
#include <hip/hip_runtime.h>
#include <hip/hip_bf16.h>
#include <cstdint>

typedef __attribute__((ext_vector_type(16))) _Float16 v16h;
typedef __attribute__((ext_vector_type(4)))  _Float16 v4h;
typedef __attribute__((ext_vector_type(8)))  float    v8f;

#define D_MODEL 1024
#define SEQ     2048
#define NHEADS  16
#define DKH     64
#define MROWS   8192   // B*S = 4*2048

// ---------------------------------------------------------------------------
// CDNA5 async global->LDS DMA (ASYNCcnt-tracked). Probe round 2 confirmed the
// builtin exists and takes (int4 AS1*, int4 AS3*, imm offset, imm cpol).
// Guarded so compile never regresses; fallback is a batched register copy.
// ---------------------------------------------------------------------------
#if defined(__gfx1250__) && __has_builtin(__builtin_amdgcn_global_load_async_to_lds_b128)
#define HAS_ASYNC_LDS 1
#else
#define HAS_ASYNC_LDS 0
#endif

#if HAS_ASYNC_LDS
typedef int v4i_gcc __attribute__((vector_size(16)));
typedef v4i_gcc __attribute__((address_space(1)))* async_gptr;
typedef v4i_gcc __attribute__((address_space(3)))* async_lptr;

__device__ inline void async_cp16(const _Float16* g, _Float16* l) {
    // generic->AS1 via 64-bit int; generic LDS ptr low 32 bits == LDS offset
    __builtin_amdgcn_global_load_async_to_lds_b128(
        (async_gptr)(uintptr_t)g,
        (async_lptr)(uint32_t)(uintptr_t)l,
        0, 0);
}
#endif

__device__ inline void async_wait() {
#if HAS_ASYNC_LDS
#if __has_builtin(__builtin_amdgcn_s_wait_asynccnt)
    __builtin_amdgcn_s_wait_asynccnt(0);
#else
    asm volatile("s_wait_asynccnt 0x0" ::: "memory");
#endif
#endif
}

// ---------------------------------------------------------------------------
// Fragment loaders (wave32); A/B use this code's contiguous k order, not the ISA table order:
//  A 16x32 f16: lane = (m&15) + 16*h ; element j:
//      k = h*16 + j
//  B 32x16 f16: lane = n + 16*h ; element j: k = h*16 + j
//  C/D 16x16 f32: lane holds (row = r + 8*h, col = n), vgpr r = 0..7
// ---------------------------------------------------------------------------
__device__ inline v16h load_frag_a(const _Float16* __restrict__ src, int ldm, int lane) {
    const int m = lane & 15, h = lane >> 4;
    v16h a;
#pragma unroll
    for (int j = 0; j < 16; ++j) {
        const int k  = h * 16 + j;
        a[j] = src[m * ldm + k];
    }
    return a;
}

// ---------------------------------------------------------------------------
// fp32 -> f16 pre-conversion (one-time, bandwidth-bound)
// ---------------------------------------------------------------------------
__global__ __launch_bounds__(256) void cvt_f16(const float* __restrict__ src,
                                               _Float16* __restrict__ dst, int n4) {
    const int i = blockIdx.x * blockDim.x + threadIdx.x;
    if (i < n4) {
        float4 v = ((const float4*)src)[i];
        v4h o;
        o[0] = (_Float16)v.x; o[1] = (_Float16)v.y;
        o[2] = (_Float16)v.z; o[3] = (_Float16)v.w;
        ((v4h*)dst)[i] = o;
    }
}

// ---------------------------------------------------------------------------
// Stage one 128x32 f16 tile pair (A from Xh rows, B from Wh rows) into LDS.
// 512 x 16B chunks per tile; 256 threads -> 2 chunks each per tile.
// ---------------------------------------------------------------------------
__device__ inline void gemm_issue_tile(
    const _Float16* __restrict__ Xh, const _Float16* __restrict__ Wh,
    _Float16* As, _Float16* Bs, int m0, int n0, int k0, int tid)
{
    constexpr int K = D_MODEL, BK = 32;
#if HAS_ASYNC_LDS
#pragma unroll
    for (int i = 0; i < 2; ++i) {
        const int c = tid + i * 256;
        const int row = c >> 2, col = (c & 3) * 8;
        async_cp16(Xh + (size_t)(m0 + row) * K + k0 + col, As + row * BK + col);
        async_cp16(Wh + (size_t)(n0 + row) * K + k0 + col, Bs + row * BK + col);
    }
#else
    uint4 ra[2], rb[2];
    int rows[2], cols[2];
#pragma unroll
    for (int i = 0; i < 2; ++i) {
        const int c = tid + i * 256;
        rows[i] = c >> 2; cols[i] = (c & 3) * 8;
        ra[i] = *(const uint4*)(Xh + (size_t)(m0 + rows[i]) * K + k0 + cols[i]);
        rb[i] = *(const uint4*)(Wh + (size_t)(n0 + rows[i]) * K + k0 + cols[i]);
    }
#pragma unroll
    for (int i = 0; i < 2; ++i) {
        *(uint4*)(As + rows[i] * BK + cols[i]) = ra[i];
        *(uint4*)(Bs + rows[i] * BK + cols[i]) = rb[i];
    }
#endif
}

// ---------------------------------------------------------------------------
// GEMM: Y = X @ W^T + bias  (Xh: [M,1024] f16, Wh: [1024,1024] f16 row-major)
// mode 0: write f16 reshaped [B,H,S,dk]   (QKV projections)
// mode 1: write f32 [M,1024]              (output projection)
// BM=BN=128, BK=32; 256 threads = 8 waves (4x2), wave tile 32x64.
// Double-buffered LDS; async DMA prefetch of tile k+1 overlaps 8 WMMAs.
// ---------------------------------------------------------------------------
__global__ __launch_bounds__(256) void gemm_wmma(
    const _Float16* __restrict__ Xh, const _Float16* __restrict__ Wh,
    const float* __restrict__ bias,
    _Float16* __restrict__ outH, float* __restrict__ outF, int mode)
{
    constexpr int BM = 128, BN = 128, BK = 32, K = D_MODEL;
    __shared__ alignas(16) _Float16 As[2][BM * BK];
    __shared__ alignas(16) _Float16 Bs[2][BN * BK];

    const int tid  = threadIdx.x;
    const int lane = tid & 31;
    const int wave = tid >> 5;
    const int wm   = wave >> 1;   // 0..3
    const int wn   = wave & 1;    // 0..1
    const int m0   = blockIdx.y * BM;
    const int n0   = blockIdx.x * BN;

    v8f zero = {0.f, 0.f, 0.f, 0.f, 0.f, 0.f, 0.f, 0.f};
    v8f acc[2][4];
#pragma unroll
    for (int i = 0; i < 2; ++i)
#pragma unroll
        for (int j = 0; j < 4; ++j) acc[i][j] = zero;

    gemm_issue_tile(Xh, Wh, As[0], Bs[0], m0, n0, 0, tid);
    async_wait();
    __syncthreads();

    int cur = 0;
    for (int k0 = 0; k0 < K; k0 += BK) {
        if (k0 + BK < K)
            gemm_issue_tile(Xh, Wh, As[cur ^ 1], Bs[cur ^ 1], m0, n0, k0 + BK, tid);

        v16h af[2];
#pragma unroll
        for (int mf = 0; mf < 2; ++mf)
            af[mf] = load_frag_a(As[cur] + (wm * 32 + mf * 16) * BK, BK, lane);

        const int n = lane & 15, h = lane >> 4;
#pragma unroll
        for (int nf = 0; nf < 4; ++nf) {
            v16h bf;
            const _Float16* bb = Bs[cur] + (wn * 64 + nf * 16 + n) * BK + h * 16;
#pragma unroll
            for (int j = 0; j < 16; ++j) bf[j] = bb[j];
#pragma unroll
            for (int mf = 0; mf < 2; ++mf)
                acc[mf][nf] = __builtin_amdgcn_wmma_f32_16x16x32_f16(
                    false, af[mf], false, bf, (short)0, acc[mf][nf], false, false);
        }
        async_wait();
        __syncthreads();
        cur ^= 1;
    }

    // Epilogue: bias + store
    const int n = lane & 15, h = lane >> 4;
#pragma unroll
    for (int mf = 0; mf < 2; ++mf) {
#pragma unroll
        for (int nf = 0; nf < 4; ++nf) {
#pragma unroll
            for (int r = 0; r < 8; ++r) {
                const int row = m0 + wm * 32 + mf * 16 + r + 8 * h;
                const int col = n0 + wn * 64 + nf * 16 + n;
                const float val = acc[mf][nf][r] + bias[col];
                if (mode == 0) {
                    const int b = row >> 11, s = row & (SEQ - 1);
                    const int hh = col >> 6, i = col & (DKH - 1);
                    outH[(((size_t)(b * NHEADS + hh) * SEQ) + s) * DKH + i] = (_Float16)val;
                } else {
                    outF[(size_t)row * D_MODEL + col] = val;
                }
            }
        }
    }
}

// ---------------------------------------------------------------------------
// Stage one 32x64 f16 K/V tile pair into LDS (256 x 16B chunks each;
// 128 threads -> 2 chunks per tile per thread).
// ---------------------------------------------------------------------------
__device__ inline void attn_issue_kv(
    const _Float16* __restrict__ Kg, const _Float16* __restrict__ Vg,
    _Float16* Ks, _Float16* Vs, size_t base, int t0, int tid)
{
#if HAS_ASYNC_LDS
#pragma unroll
    for (int i = 0; i < 2; ++i) {
        const int c = (tid + i * 128) * 8;
        async_cp16(Kg + base + (size_t)t0 * DKH + c, Ks + c);
        async_cp16(Vg + base + (size_t)t0 * DKH + c, Vs + c);
    }
#else
    uint4 rk[2], rv[2];
#pragma unroll
    for (int i = 0; i < 2; ++i) {
        const int c = (tid + i * 128) * 8;
        rk[i] = *(const uint4*)(Kg + base + (size_t)t0 * DKH + c);
        rv[i] = *(const uint4*)(Vg + base + (size_t)t0 * DKH + c);
    }
#pragma unroll
    for (int i = 0; i < 2; ++i) {
        const int c = (tid + i * 128) * 8;
        *(uint4*)(Ks + c) = rk[i];
        *(uint4*)(Vs + c) = rv[i];
    }
#endif
}

// ---------------------------------------------------------------------------
// Flash attention: one block per (b,h, 64-query tile). 128 threads = 4 waves,
// each wave owns 16 query rows. Double-buffered 32-key K/V tiles in LDS.
// ---------------------------------------------------------------------------
__global__ __launch_bounds__(128) void attn_flash(
    const _Float16* __restrict__ Q, const _Float16* __restrict__ Kg,
    const _Float16* __restrict__ Vg, _Float16* __restrict__ headOut)
{
    __shared__ alignas(16) _Float16 Qs[64 * DKH];
    __shared__ alignas(16) _Float16 Ks[2][32 * DKH];
    __shared__ alignas(16) _Float16 Vs[2][32 * DKH];
    __shared__ alignas(16) _Float16 Ps[4 * 16 * 32];   // per-wave P stash

    const int tid  = threadIdx.x;
    const int lane = tid & 31;
    const int wave = tid >> 5;
    const int bh   = blockIdx.x >> 5;    // 32 q-tiles per (b,h)
    const int q0   = (blockIdx.x & 31) * 64;
    const size_t base = (size_t)bh * SEQ * DKH;

    // Stage the 64x64 Q tile (512 chunks) and the first K/V tile together.
#if HAS_ASYNC_LDS
#pragma unroll
    for (int i = 0; i < 4; ++i) {
        const int c = (tid + i * 128) * 8;
        async_cp16(Q + base + (size_t)q0 * DKH + c, Qs + c);
    }
#else
    {
        const uint4* src = (const uint4*)(Q + base + (size_t)q0 * DKH);
        uint4* dst = (uint4*)Qs;
        uint4 r[4];
#pragma unroll
        for (int i = 0; i < 4; ++i) r[i] = src[tid + i * 128];
#pragma unroll
        for (int i = 0; i < 4; ++i) dst[tid + i * 128] = r[i];
    }
#endif
    attn_issue_kv(Kg, Vg, Ks[0], Vs[0], base, 0, tid);
    async_wait();
    __syncthreads();

    v16h aQ[2];
#pragma unroll
    for (int kk = 0; kk < 2; ++kk)
        aQ[kk] = load_frag_a(Qs + (wave * 16) * DKH + kk * 32, DKH, lane);

    v8f zero = {0.f, 0.f, 0.f, 0.f, 0.f, 0.f, 0.f, 0.f};
    v8f of[4];
    float mrow[8], lrow[8];
#pragma unroll
    for (int i = 0; i < 4; ++i) of[i] = zero;
#pragma unroll
    for (int r = 0; r < 8; ++r) { mrow[r] = -1e30f; lrow[r] = 0.f; }

    const int n = lane & 15, h = lane >> 4;
    int cur = 0;

    for (int t0 = 0; t0 < SEQ; t0 += 32) {
        if (t0 + 32 < SEQ)
            attn_issue_kv(Kg, Vg, Ks[cur ^ 1], Vs[cur ^ 1], base, t0 + 32, tid);

        // S = Q K^T for a 16x32 stripe per wave (two 16x16 C tiles)
        v8f sf[2];
#pragma unroll
        for (int nt = 0; nt < 2; ++nt) {
            v8f s = zero;
#pragma unroll
            for (int kk = 0; kk < 2; ++kk) {
                v16h bK;
                const _Float16* kb = Ks[cur] + (nt * 16 + n) * DKH + kk * 32 + h * 16;
#pragma unroll
                for (int j = 0; j < 16; ++j) bK[j] = kb[j];
                s = __builtin_amdgcn_wmma_f32_16x16x32_f16(
                        false, aQ[kk], false, bK, (short)0, s, false, false);
            }
            sf[nt] = s;
        }

        // Online softmax: rows live across the 16-lane half-groups.
#pragma unroll
        for (int r = 0; r < 8; ++r) {
            float a = sf[0][r] * 0.125f;       // 1/sqrt(64)
            float b = sf[1][r] * 0.125f;
            float mx = fmaxf(a, b);
#pragma unroll
            for (int off = 1; off < 16; off <<= 1)
                mx = fmaxf(mx, __shfl_xor(mx, off, 16));
            const float mnew = fmaxf(mrow[r], mx);
            const float corr = __expf(mrow[r] - mnew);
            const float p0 = __expf(a - mnew);
            const float p1 = __expf(b - mnew);
            float rs = p0 + p1;
#pragma unroll
            for (int off = 1; off < 16; off <<= 1)
                rs += __shfl_xor(rs, off, 16);
            lrow[r] = lrow[r] * corr + rs;
            mrow[r] = mnew;
            sf[0][r] = p0;
            sf[1][r] = p1;
#pragma unroll
            for (int nf = 0; nf < 4; ++nf) of[nf][r] *= corr;
        }

        // Repack P (C-layout) -> A-layout through per-wave LDS (same-wave DS
        // ops are in-order, no barrier needed).
        _Float16* pw = Ps + wave * 512;
#pragma unroll
        for (int nt = 0; nt < 2; ++nt)
#pragma unroll
            for (int r = 0; r < 8; ++r)
                pw[(r + 8 * h) * 32 + nt * 16 + n] = (_Float16)sf[nt][r];

        v16h aP = load_frag_a(pw, 32, lane);

        // O += P V
#pragma unroll
        for (int nf = 0; nf < 4; ++nf) {
            v16h bV;
#pragma unroll
            for (int j = 0; j < 16; ++j)
                bV[j] = Vs[cur][(h * 16 + j) * DKH + nf * 16 + n];
            of[nf] = __builtin_amdgcn_wmma_f32_16x16x32_f16(
                         false, aP, false, bV, (short)0, of[nf], false, false);
        }

        async_wait();
        __syncthreads();
        cur ^= 1;
    }

    const int b  = bh >> 4;
    const int hh = bh & 15;
#pragma unroll
    for (int r = 0; r < 8; ++r) {
        const float inv = 1.0f / lrow[r];
        const int row = q0 + wave * 16 + r + 8 * h;
#pragma unroll
        for (int nf = 0; nf < 4; ++nf)
            headOut[(size_t)(b * SEQ + row) * D_MODEL + hh * DKH + nf * 16 + n] =
                (_Float16)(of[nf][r] * inv);
    }
}

// ---------------------------------------------------------------------------
// Fused LayerNorm + residual: out = x + (o - mu)*rstd*gamma + beta
// One 256-thread block per row of 1024 (float4 per thread).
// ---------------------------------------------------------------------------
__global__ __launch_bounds__(256) void ln_residual(
    const float* __restrict__ x, const float* __restrict__ o,
    const float* __restrict__ gamma, const float* __restrict__ beta,
    float* __restrict__ out)
{
    const int row = blockIdx.x, tid = threadIdx.x;
    const size_t base = (size_t)row * D_MODEL;

    float4 ov = ((const float4*)(o + base))[tid];
    float s = ov.x + ov.y + ov.z + ov.w;
    float q = ov.x * ov.x + ov.y * ov.y + ov.z * ov.z + ov.w * ov.w;
#pragma unroll
    for (int off = 1; off < 32; off <<= 1) {
        s += __shfl_xor(s, off, 32);
        q += __shfl_xor(q, off, 32);
    }
    __shared__ float ss[8], sq[8];
    if ((tid & 31) == 0) { ss[tid >> 5] = s; sq[tid >> 5] = q; }
    __syncthreads();
    float tot = 0.f, totq = 0.f;
#pragma unroll
    for (int i = 0; i < 8; ++i) { tot += ss[i]; totq += sq[i]; }
    const float mu   = tot * (1.f / D_MODEL);
    const float var  = totq * (1.f / D_MODEL) - mu * mu;
    const float rstd = rsqrtf(var + 1e-5f);

    float4 xv = ((const float4*)(x + base))[tid];
    float4 gv = ((const float4*)gamma)[tid];
    float4 bv = ((const float4*)beta)[tid];
    float4 r;
    r.x = xv.x + (ov.x - mu) * rstd * gv.x + bv.x;
    r.y = xv.y + (ov.y - mu) * rstd * gv.y + bv.y;
    r.z = xv.z + (ov.z - mu) * rstd * gv.z + bv.z;
    r.w = xv.w + (ov.w - mu) * rstd * gv.w + bv.w;
    ((float4*)(out + base))[tid] = r;
}

// ---------------------------------------------------------------------------
extern "C" void kernel_launch(void* const* d_in, const int* in_sizes, int n_in,
                              void* d_out, int out_size, void* d_ws, size_t ws_size,
                              hipStream_t stream) {
    const float* x     = (const float*)d_in[0];
    const float* Wq    = (const float*)d_in[1];
    const float* bq    = (const float*)d_in[2];
    const float* Wk    = (const float*)d_in[3];
    const float* bk    = (const float*)d_in[4];
    const float* Wv    = (const float*)d_in[5];
    const float* bv    = (const float*)d_in[6];
    const float* Wo    = (const float*)d_in[7];
    const float* bo    = (const float*)d_in[8];
    const float* gamma = (const float*)d_in[9];
    const float* beta  = (const float*)d_in[10];

    char* ws = (char*)d_ws;
    const size_t XH_B = (size_t)MROWS * D_MODEL * 2;       // 16 MiB
    const size_t WH_B = (size_t)D_MODEL * D_MODEL * 2;     //  2 MiB
    _Float16* Xh  = (_Float16*)(ws);
    _Float16* Wqh = (_Float16*)(ws + XH_B);
    _Float16* Wkh = (_Float16*)(ws + XH_B + WH_B);
    _Float16* Wvh = (_Float16*)(ws + XH_B + 2 * WH_B);
    _Float16* Woh = (_Float16*)(ws + XH_B + 3 * WH_B);
    _Float16* qws = (_Float16*)(ws + XH_B + 4 * WH_B);
    _Float16* kws = (_Float16*)(ws + 2 * XH_B + 4 * WH_B);
    _Float16* vws = (_Float16*)(ws + 3 * XH_B + 4 * WH_B);
    _Float16* hdh = (_Float16*)(ws + 4 * XH_B + 4 * WH_B);
    float*    oprj = (float*)  (ws + 5 * XH_B + 4 * WH_B); // 32 MiB

    // one-time fp32 -> f16 conversion of activations + weights
    cvt_f16<<<(MROWS * D_MODEL / 4) / 256, 256, 0, stream>>>(x,  Xh,  MROWS * D_MODEL / 4);
    cvt_f16<<<(D_MODEL * D_MODEL / 4) / 256, 256, 0, stream>>>(Wq, Wqh, D_MODEL * D_MODEL / 4);
    cvt_f16<<<(D_MODEL * D_MODEL / 4) / 256, 256, 0, stream>>>(Wk, Wkh, D_MODEL * D_MODEL / 4);
    cvt_f16<<<(D_MODEL * D_MODEL / 4) / 256, 256, 0, stream>>>(Wv, Wvh, D_MODEL * D_MODEL / 4);
    cvt_f16<<<(D_MODEL * D_MODEL / 4) / 256, 256, 0, stream>>>(Wo, Woh, D_MODEL * D_MODEL / 4);

    dim3 gg(D_MODEL / 128, MROWS / 128);   // (8, 64)
    gemm_wmma<<<gg, 256, 0, stream>>>(Xh, Wqh, bq, qws, nullptr, 0);
    gemm_wmma<<<gg, 256, 0, stream>>>(Xh, Wkh, bk, kws, nullptr, 0);
    gemm_wmma<<<gg, 256, 0, stream>>>(Xh, Wvh, bv, vws, nullptr, 0);

    attn_flash<<<4 * NHEADS * (SEQ / 64), 128, 0, stream>>>(qws, kws, vws, hdh);

    gemm_wmma<<<gg, 256, 0, stream>>>(hdh, Woh, bo, nullptr, oprj, 1);

    ln_residual<<<MROWS, 256, 0, stream>>>(x, oprj, gamma, beta, (float*)d_out);
}
